// MultiHeadAttentionLayer_48558900249388
// MI455X (gfx1250) — matrix operand order repaired and checked by reading, not compiled, not run on hardware
//
#include <hip/hip_runtime.h>

// ---------------------------------------------------------------------------
// MI455X (gfx1250) implementation, round 2.
//  prep_h   : h f32 -> bf16, zero-padded to Npad rows (no guards in GEMM)
//  prep_w   : Wq/Wk/Wv f32 [k][n] -> bf16 transposed Wt[z][n][k]
//  qkv_gemm : bf16 WMMA GEMM, double-buffered async global->LDS staging
//             (global_load_async_to_lds_b128 + s_wait_asynccnt), fused 2D
//             RoPE epilogue on Q,K; Q/K/V stored bf16 (halves gather bytes)
//  edge_attn: wave-per-node softmax aggregation (dst = repeat(arange(N),16))
// Workspace: hb | Wt(3) | Qb | Kb | Vb  (all bf16, Npad rows)
// ---------------------------------------------------------------------------

typedef __bf16 v8bf  __attribute__((ext_vector_type(8)));
typedef __bf16 v16bf __attribute__((ext_vector_type(16)));
typedef float  v8f   __attribute__((ext_vector_type(8)));

#define CDIM 256
#define BM 128
#define BN 128
#define BK 32
#define LDSS 40           // ushort row stride: 80B (16B aligned, bank-spread)

__device__ __forceinline__ unsigned short f2bf(float f) {
    union { float f; unsigned u; } v; v.f = f;
    unsigned r = v.u + 0x7FFFu + ((v.u >> 16) & 1u);   // RNE
    return (unsigned short)(r >> 16);
}

// Async global->LDS 16B copy (CDNA5, tracked by ASYNCcnt). GV addressing.
__device__ __forceinline__ void async_ld16(unsigned lds_addr, unsigned long long gaddr) {
    asm volatile("global_load_async_to_lds_b128 %0, %1, off"
                 :: "v"(lds_addr), "v"(gaddr) : "memory");
}
__device__ __forceinline__ void wait_async0() {
    asm volatile("s_wait_asynccnt 0x0" ::: "memory");
}

// ---------------- prep kernels (one-time, tiny) ----------------------------
__global__ __launch_bounds__(256)
void prep_h(const float* __restrict__ h, __bf16* __restrict__ hb, int N) {
    int t = blockIdx.x * 256 + threadIdx.x;          // one 8-elem chunk each
    size_t base = (size_t)t * 8;
    int row = (int)(base >> 8);
    unsigned short u[8];
    if (row < N) {
        float4 a = *(const float4*)(h + base);
        float4 b = *(const float4*)(h + base + 4);
        u[0] = f2bf(a.x); u[1] = f2bf(a.y); u[2] = f2bf(a.z); u[3] = f2bf(a.w);
        u[4] = f2bf(b.x); u[5] = f2bf(b.y); u[6] = f2bf(b.z); u[7] = f2bf(b.w);
    } else {
        for (int i = 0; i < 8; ++i) u[i] = 0;
    }
    uint4 pk;
    pk.x = (unsigned)u[0] | ((unsigned)u[1] << 16);
    pk.y = (unsigned)u[2] | ((unsigned)u[3] << 16);
    pk.z = (unsigned)u[4] | ((unsigned)u[5] << 16);
    pk.w = (unsigned)u[6] | ((unsigned)u[7] << 16);
    *(uint4*)((unsigned short*)hb + base) = pk;
}

__global__ __launch_bounds__(256)
void prep_w(const float* __restrict__ Wq, const float* __restrict__ Wk,
            const float* __restrict__ Wv, __bf16* __restrict__ Wt) {
    const int z = blockIdx.y;
    const float* __restrict__ W = (z == 0) ? Wq : (z == 1) ? Wk : Wv;
    int o = blockIdx.x * 256 + threadIdx.x;          // input-linear, coalesced
    int k = o >> 8, n = o & 255;
    ((unsigned short*)Wt)[(size_t)z * CDIM * CDIM + (size_t)n * CDIM + k] = f2bf(W[o]);
}

// ---------------- fused QKV GEMM + RoPE ------------------------------------
__global__ __launch_bounds__(256)
void qkv_rope_gemm(const __bf16* __restrict__ hb,
                   const __bf16* __restrict__ Wt,
                   const float* __restrict__ posp,
                   __bf16* __restrict__ Qb,
                   __bf16* __restrict__ Kb,
                   __bf16* __restrict__ Vb,
                   int N) {
    __shared__ unsigned short Abuf[2][BM * LDSS];
    __shared__ unsigned short Bbuf[2][BN * LDSS];

    const int z = blockIdx.z;
    const __bf16* __restrict__ wt = Wt + (size_t)z * CDIM * CDIM;
    __bf16* __restrict__ outp = (z == 0) ? Qb : (z == 1) ? Kb : Vb;
    const bool do_rope = (z < 2);

    const int m0 = blockIdx.x * BM;
    const int n0 = blockIdx.y * BN;

    const int tid    = threadIdx.x;
    const int lane   = tid & 31;
    const int wid    = tid >> 5;
    const int wave_m = wid & 3;       // 4 waves over M
    const int wave_n = wid >> 2;      // 2 waves over N
    const int lm     = lane & 15;
    const bool hiL   = (lane >= 16);

    v8f acc[2][4];
    for (int mt = 0; mt < 2; ++mt)
        for (int nt = 0; nt < 4; ++nt)
            acc[mt][nt] = (v8f)0.0f;

    // stage tile k0 into buffer b: 128 rows x 64B each for A and B panels
    auto stage = [&](int b, int k0) {
        for (int i = 0; i < 2; ++i) {
            int lin = tid + i * 256;              // 512 16B chunks per panel
            int row = lin >> 2;
            int c   = lin & 3;
            async_ld16((unsigned)(uintptr_t)&Abuf[b][row * LDSS + c * 8],
                       (unsigned long long)(uintptr_t)(hb + (size_t)(m0 + row) * CDIM + k0 + c * 8));
            async_ld16((unsigned)(uintptr_t)&Bbuf[b][row * LDSS + c * 8],
                       (unsigned long long)(uintptr_t)(wt + (size_t)(n0 + row) * CDIM + k0 + c * 8));
        }
    };

    stage(0, 0);
    wait_async0();
    __syncthreads();

    for (int kt = 0; kt < CDIM / BK; ++kt) {
        const int cur = kt & 1;
        if (kt < (CDIM / BK) - 1)
            stage(cur ^ 1, (kt + 1) * BK);        // prefetch next tile (async)

        // fragment loads per ISA 16-bit A/B VGPR layouts
        v16bf afrag[2];
        for (int mt = 0; mt < 2; ++mt) {
            int rowl = wave_m * 32 + mt * 16 + lm;
            int o0 = hiL ? 8 : 0;
            v8bf a_lo = *(const v8bf*)&Abuf[cur][rowl * LDSS + o0];
            v8bf a_hi = *(const v8bf*)&Abuf[cur][rowl * LDSS + 16 + o0];
            afrag[mt] = __builtin_shufflevector(a_lo, a_hi,
                0,1,2,3,4,5,6,7,8,9,10,11,12,13,14,15);
        }
        v16bf bfrag[4];
        for (int nt = 0; nt < 4; ++nt) {
            int nl = wave_n * 64 + nt * 16 + lm;
            int ob = hiL ? 8 : 0;
            v8bf b_lo = *(const v8bf*)&Bbuf[cur][nl * LDSS + ob];
            v8bf b_hi = *(const v8bf*)&Bbuf[cur][nl * LDSS + ob + 16];
            bfrag[nt] = __builtin_shufflevector(b_lo, b_hi,
                0,1,2,3,4,5,6,7,8,9,10,11,12,13,14,15);
        }
        for (int mt = 0; mt < 2; ++mt)
            for (int nt = 0; nt < 4; ++nt)
                acc[mt][nt] = __builtin_amdgcn_wmma_f32_16x16x32_bf16(
                    false, afrag[mt], false, bfrag[nt],
                    (short)0, acc[mt][nt], false, false);

        if (kt < (CDIM / BK) - 1) {
            wait_async0();                        // next buffer resident
            __syncthreads();                      // all waves done with cur
        }
    }

    // ---- epilogue: 2D RoPE (Q,K only) + bf16 store (buffers are padded) ----
    const float invf = __expf(-1.1512925464970229f * (float)(lane & 7));
    const float sgn  = (lane & 8) ? 1.0f : -1.0f;

    for (int mt = 0; mt < 2; ++mt) {
        for (int nt = 0; nt < 4; ++nt) {
            const int nbase = n0 + wave_n * 64 + nt * 16;
            const int col   = nbase + lm;
            const int axis  = (nbase >> 4) & 1;
            for (int j = 0; j < 8; ++j) {
                int row = m0 + wave_m * 32 + mt * 16 + j + (hiL ? 8 : 0);
                float t = acc[mt][nt][j];
                float outv = t;
                if (do_rope) {
                    float prt = __shfl_xor(t, 8, 32);
                    float pv  = (row < N) ? posp[(size_t)row * 2 + axis] : 0.f;
                    float fr  = pv * 64.0f * invf;
                    float s, c;
                    __sincosf(fr, &s, &c);
                    outv = t * c + sgn * prt * s;
                }
                ((unsigned short*)outp)[(size_t)row * CDIM + col] = f2bf(outv);
            }
        }
    }
}

// ---------------- edge attention (wave per node, bf16 gathers) -------------
__global__ __launch_bounds__(256)
void edge_attn(const __bf16* __restrict__ Q,
               const __bf16* __restrict__ K,
               const __bf16* __restrict__ V,
               const int* __restrict__ src,
               float* __restrict__ out,
               int N, int deg) {
    const int wid  = threadIdx.x >> 5;
    const int lane = threadIdx.x & 31;
    const int node = blockIdx.x * 8 + wid;
    if (node >= N) return;

    const int* __restrict__ s = src + (size_t)node * deg;
    const float rscale = 0.17677669529663689f;   // 1/sqrt(32)

    for (int h = 0; h < 8; ++h) {
        const size_t qoff = (size_t)node * CDIM + h * 32 + lane;
        const float q = (float)Q[qoff];
        float o = 0.f, zden = 0.f;
        for (int e = 0; e < deg; ++e) {
            int sv = s[e];
            size_t off = (size_t)sv * CDIM + h * 32 + lane;
            float p = (float)K[off] * q;
            p += __shfl_xor(p, 16, 32);
            p += __shfl_xor(p,  8, 32);
            p += __shfl_xor(p,  4, 32);
            p += __shfl_xor(p,  2, 32);
            p += __shfl_xor(p,  1, 32);
            float sc = __expf(fminf(fmaxf(p * rscale, -5.f), 5.f));
            o += sc * (float)V[off];
            zden += sc;
        }
        out[qoff] = o / zden;
    }
}

extern "C" void kernel_launch(void* const* d_in, const int* in_sizes, int n_in,
                              void* d_out, int out_size, void* d_ws, size_t ws_size,
                              hipStream_t stream) {
    const float* h   = (const float*)d_in[0];
    const float* pos = (const float*)d_in[1];
    const float* Wq  = (const float*)d_in[2];
    const float* Wk  = (const float*)d_in[3];
    const float* Wv  = (const float*)d_in[4];
    const int*   src = (const int*)d_in[5];

    const int N    = in_sizes[1] / 2;             // pos is [N,2]
    const int Npad = ((N + BM - 1) / BM) * BM;
    const int E    = in_sizes[5];
    const int deg  = E / N;

    const size_t hbn = (size_t)Npad * CDIM;       // elems per bf16 matrix
    __bf16* hb = (__bf16*)d_ws;
    __bf16* Wt = hb + hbn;
    __bf16* Qb = Wt + (size_t)3 * CDIM * CDIM;
    __bf16* Kb = Qb + hbn;
    __bf16* Vb = Kb + hbn;

    prep_h<<<Npad / 8, 256, 0, stream>>>(h, hb, N);
    prep_w<<<dim3(CDIM, 3), 256, 0, stream>>>(Wq, Wk, Wv, Wt);

    dim3 g1(Npad / BM, CDIM / BN, 3);
    qkv_rope_gemm<<<g1, 256, 0, stream>>>(hb, Wt, pos, Qb, Kb, Vb, N);

    dim3 g2((N + 7) / 8);
    edge_attn<<<g2, 256, 0, stream>>>(Qb, Kb, Vb, src, (float*)d_out, N, deg);
}
